// DPF_32109175505662
// MI455X (gfx1250) — matrix operand order repaired and checked by reading, not compiled, not run on hardware
//
#include <hip/hip_runtime.h>
#include <math.h>

typedef _Float16 half_t;
typedef __attribute__((ext_vector_type(16))) _Float16 v16h;
typedef __attribute__((ext_vector_type(8)))  float    v8f;

constexpr int NB = 32, NT = 50, NP = 4096, NH = 128, NOBS = 1728;
constexpr float STDV = 0.1f;

// ---- output layout (floats, concatenated in return order) ----
constexpr size_t PL_OFF = 0;                                   // [B,T,P,3]
constexpr size_t PR_OFF = (size_t)NB * NT * NP * 3;            // [B,T,P]
constexpr size_t IL_OFF = PR_OFF + (size_t)NB * NT * NP;       // [B,P,T]
constexpr size_t LL_OFF = IL_OFF + (size_t)NB * NP * NT;       // scalar

// ---- workspace layout (float offsets) ----
constexpr size_t WS_LOGW = 0;                         // [B*P]
constexpr size_t WS_CUM  = WS_LOGW + (size_t)NB * NP; // [B*P]
constexpr size_t WS_BIAS = WS_CUM + (size_t)NB * NP;  // [B*T*3]
constexpr size_t WS_ENC  = WS_BIAS + NB * NT * 3;     // [B*T*H]
constexpr size_t WS_ENRM = WS_ENC + (size_t)NB * NT * NH; // [B*T]
constexpr size_t WS_ESS  = WS_ENRM + NB * NT;         // [B]
constexpr size_t WS_ACC  = WS_ESS + 32;               // [B]
constexpr size_t WS_W2H  = WS_ACC + 32;               // half[H*H] (32B aligned)
// h_enc [B*T*H] aliases WS_LOGW..   (used only before k_init)

// ------------------------------------------------------------------
__global__ void k_cvt_w2(const float* __restrict__ w, half_t* __restrict__ wh) {
  int i = blockIdx.x * blockDim.x + threadIdx.x;
  if (i < NH * NH) wh[i] = (half_t)w[i];
}

// obs encoder layer 1: [B*T,1728]@[1728,128] + relu
__global__ void k_enc1(const float* __restrict__ obs, const float* __restrict__ W1,
                       const float* __restrict__ b1, float* __restrict__ henc) {
  int idx = blockIdx.x * blockDim.x + threadIdx.x;
  if (idx >= NB * NT * NH) return;
  int row = idx >> 7, j = idx & 127;
  const float* o = obs + (size_t)row * NOBS;
  float acc = b1[j];
  for (int k = 0; k < NOBS; ++k) acc += o[k] * W1[k * NH + j];
  henc[idx] = fmaxf(acc, 0.f);
}

// obs encoder layer 2: [B*T,128]@[128,128]
__global__ void k_enc2(const float* __restrict__ henc, const float* __restrict__ W2,
                       const float* __restrict__ b2, float* __restrict__ enc) {
  int idx = blockIdx.x * blockDim.x + threadIdx.x;
  if (idx >= NB * NT * NH) return;
  int row = idx >> 7, j = idx & 127;
  const float* h = henc + (size_t)row * NH;
  float acc = b2[j];
  for (int k = 0; k < NH; ++k) acc += h[k] * W2[k * NH + j];
  enc[idx] = acc;
}

__global__ void k_norm(const float* __restrict__ enc, float* __restrict__ nrm) {
  int i = blockIdx.x * blockDim.x + threadIdx.x;
  if (i >= NB * NT) return;
  const float* e = enc + (size_t)i * NH;
  float s = 0.f;
  for (int k = 0; k < NH; ++k) s += e[k] * e[k];
  nrm[i] = sqrtf(s);
}

// learned action-conditioned bias: mlp2(action) 3->32->3
__global__ void k_bias(const float* __restrict__ act, const float* __restrict__ W1,
                       const float* __restrict__ b1, const float* __restrict__ W2,
                       const float* __restrict__ b2, float* __restrict__ bias) {
  int i = blockIdx.x * blockDim.x + threadIdx.x;
  if (i >= NB * NT) return;
  float a0 = act[i * 3], a1 = act[i * 3 + 1], a2 = act[i * 3 + 2];
  float hh[32];
  #pragma unroll
  for (int k = 0; k < 32; ++k)
    hh[k] = fmaxf(b1[k] + a0 * W1[k] + a1 * W1[32 + k] + a2 * W1[64 + k], 0.f);
  #pragma unroll
  for (int j = 0; j < 3; ++j) {
    float acc = b2[j];
    #pragma unroll
    for (int k = 0; k < 32; ++k) acc += hh[k] * W2[k * 3 + j];
    bias[i * 3 + j] = acc;
  }
}

// init: particles0 = state[:,0] + noise, logw = log_mu; base indices; zero acc
__global__ void k_init(const float* __restrict__ state, const float* __restrict__ inz,
                       float* __restrict__ out, float* __restrict__ logw,
                       float* __restrict__ obs_acc) {
  int idx = blockIdx.x * blockDim.x + threadIdx.x;
  if (idx >= NB * NP) return;
  int b = idx >> 12, p = idx & (NP - 1);
  const float* st = state + (size_t)b * NT * 3;  // state[b,0,:]
  const float LOGC = logf(STDV * 2.5066282746310002f);
  float lm = 0.f;
  #pragma unroll
  for (int k = 0; k < 3; ++k) {
    float nz = inz[(size_t)idx * 3 + k];
    out[PL_OFF + ((size_t)(b * NT) * NP + p) * 3 + k] = st[k] + nz * STDV;
    lm += -0.5f * nz * nz - LOGC;
  }
  logw[idx] = lm;
  out[IL_OFF + (size_t)(b * NP + p) * NT] = (float)(p + NP * b);
  if (idx < NB) obs_acc[idx] = 0.f;
}

// ---- measurement: particle encoder (WMMA) + cosine, logw += lki ----
// LDS byte offsets within dynamic smem
constexpr int HS = 132;                       // h row stride (halfs)
constexpr int ES = 132;                       // ep row stride (floats)
constexpr int LDS_H_B  = 128 * HS * 2;        // 33792
constexpr int LDS_EP_B = 128 * ES * 4;        // 67584
constexpr int LDS_XS_B = 128 * 4 * 4;         // 2048
constexpr int LDS_EO_B = 128 * 4;             // 512
constexpr int LDS_W2_OFF = LDS_H_B + LDS_EP_B + LDS_XS_B + LDS_EO_B; // 103936 (32B aligned)
constexpr int LDS_TOTAL  = LDS_W2_OFF + NH * NH * 2;                 // + 32768

__global__ __launch_bounds__(256)
void k_measure(const float* __restrict__ out, int t,
               const float* __restrict__ enc_all, const float* __restrict__ enc_norm,
               const float* __restrict__ W1, const float* __restrict__ b1,
               const half_t* __restrict__ W2h, const float* __restrict__ b2,
               float* __restrict__ logw) {
  const int b = blockIdx.x >> 5;
  const int chunk = blockIdx.x & 31;       // 32 chunks of 128 particles
  const int tid = threadIdx.x;
  const int lane = tid & 31;
  const int wave = tid >> 5;               // 8 waves

  extern __shared__ char smem[];
  half_t* h_lds  = (half_t*)smem;                         // 128*HS halfs
  float*  ep_lds = (float*)(smem + LDS_H_B);              // 128*ES floats
  float*  xs_lds = (float*)(smem + LDS_H_B + LDS_EP_B);   // 128*4
  float*  eo_lds = xs_lds + 128 * 4;                      // 128
  half_t* w2_lds = (half_t*)(smem + LDS_W2_OFF);          // 128*128 halfs

  // --- async DMA: stage W_pe2 (f16, 32KB) global -> LDS; overlaps layer1 ---
  {
    // low 32 bits of a generic shared-space address == LDS byte address
    unsigned dbase = (unsigned)(size_t)(void*)w2_lds + (unsigned)tid * 16u;
    const half_t* gsrc = W2h + tid * 8;     // 8 halfs = 16B per thread per iter
    #pragma unroll
    for (int it = 0; it < 8; ++it) {
      unsigned d = dbase + (unsigned)(it * 4096);
      const half_t* g = gsrc + it * 2048;
      asm volatile("global_load_async_to_lds_b128 %0, %1, off"
                   :: "v"(d), "v"(g) : "memory");
    }
  }

  const float* pslice = out + PL_OFF + ((size_t)(b * NT + t) * NP + chunk * 128) * 3;
  if (tid < 128) {
    float x = pslice[tid * 3], y = pslice[tid * 3 + 1], th = pslice[tid * 3 + 2];
    xs_lds[tid * 4 + 0] = x;
    xs_lds[tid * 4 + 1] = y;
    xs_lds[tid * 4 + 2] = cosf(th);
    xs_lds[tid * 4 + 3] = sinf(th);
    eo_lds[tid] = enc_all[(size_t)(b * NT + t) * NH + tid];
  }
  __syncthreads();

  // layer1: h = relu(xs @ W1 + b1), K=4 (VALU), store f16 in LDS
  for (int it = 0; it < 64; ++it) {
    int idx = tid + 256 * it;
    int r = idx >> 7, c = idx & 127;
    float acc = b1[c];
    #pragma unroll
    for (int k = 0; k < 4; ++k) acc += xs_lds[r * 4 + k] * W1[k * NH + c];
    h_lds[r * HS + c] = (half_t)fmaxf(acc, 0.f);
  }
  // wait for W2 DMA (per-wave) then block barrier
  asm volatile("s_wait_asynccnt 0x0" ::: "memory");
  __syncthreads();

  // layer2 GEMM via WMMA: wave owns one 16-row M-tile; 8 N-tiles; K=128 (4 steps)
  const int mt = wave;
  const int mrow = mt * 16 + (lane & 15);
  v16h afr[4];
  #pragma unroll
  for (int kt = 0; kt < 4; ++kt) {
    #pragma unroll
    for (int v = 0; v < 8; ++v) {
      int klo = kt * 32 + 2 * v + 8 * ((v >> 2) + (lane >> 4));
      afr[kt][2 * v]     = h_lds[mrow * HS + klo];
      afr[kt][2 * v + 1] = h_lds[mrow * HS + klo + 1];
    }
  }
  for (int nt = 0; nt < 8; ++nt) {
    v8f acc = {};
    #pragma unroll
    for (int kt = 0; kt < 4; ++kt) {
      v16h bfr;
      #pragma unroll
      for (int v = 0; v < 8; ++v) {
        int klo = kt * 32 + 2 * v + 8 * ((v >> 2) + (lane >> 4));
        bfr[2 * v]     = w2_lds[(size_t)klo * NH + nt * 16 + (lane & 15)];
        bfr[2 * v + 1] = w2_lds[(size_t)(klo + 1) * NH + nt * 16 + (lane & 15)];
      }
      acc = __builtin_amdgcn_wmma_f32_16x16x32_f16(false, afr[kt], false, bfr,
                                                   (short)0, acc, false, false);
    }
    #pragma unroll
    for (int v = 0; v < 8; ++v) {
      int row = mt * 16 + v + 8 * (lane >> 4);
      int col = nt * 16 + (lane & 15);
      ep_lds[row * ES + col] = acc[v] + b2[col];
    }
  }
  __syncthreads();

  // cosine: 2 threads per particle, lane-pair combine
  int pl = tid >> 1, hf = tid & 1;
  const float* er = ep_lds + pl * ES + hf * 64;
  const float* eo = eo_lds + hf * 64;
  float num = 0.f, den = 0.f;
  for (int n = 0; n < 64; ++n) { float v = er[n]; num += eo[n] * v; den += v * v; }
  num += __shfl_xor(num, 1);
  den += __shfl_xor(den, 1);
  if (hf == 0) {
    float on = enc_norm[b * NT + t];
    float lki = num / (on * sqrtf(den) + 1e-8f);
    logw[(size_t)b * NP + chunk * 128 + pl] += lki;
  }
}

// softmax over P per batch; also accumulate sum(logw) into obs_acc[b]
__global__ __launch_bounds__(256)
void k_softmax(const float* __restrict__ logw, float* __restrict__ out,
               float* __restrict__ obs_acc, int t) {
  int b = blockIdx.x, tid = threadIdx.x;
  __shared__ float red[256];
  const float* lw = logw + (size_t)b * NP;
  float loc[16];
  float m = -1e30f;
  for (int i = 0; i < 16; ++i) { loc[i] = lw[tid * 16 + i]; m = fmaxf(m, loc[i]); }
  red[tid] = m; __syncthreads();
  for (int off = 128; off > 0; off >>= 1) {
    if (tid < off) red[tid] = fmaxf(red[tid], red[tid + off]);
    __syncthreads();
  }
  m = red[0]; __syncthreads();
  float sl = 0.f;
  for (int i = 0; i < 16; ++i) sl += loc[i];
  float se = 0.f;
  for (int i = 0; i < 16; ++i) { loc[i] = expf(loc[i] - m); se += loc[i]; }
  red[tid] = se; __syncthreads();
  for (int off = 128; off > 0; off >>= 1) {
    if (tid < off) red[tid] += red[tid + off];
    __syncthreads();
  }
  se = red[0]; __syncthreads();
  red[tid] = sl; __syncthreads();
  for (int off = 128; off > 0; off >>= 1) {
    if (tid < off) red[tid] += red[tid + off];
    __syncthreads();
  }
  sl = red[0];
  float inv = 1.f / se;
  float* pr = out + PR_OFF + (size_t)(b * NT + t) * NP;
  for (int i = 0; i < 16; ++i) pr[tid * 16 + i] = loc[i] * inv;
  if (tid == 0) obs_acc[b] += sl;
}

// cumsum of probs[t-1] per batch + per-batch 1/sum(probs^2)
__global__ __launch_bounds__(256)
void k_cum(const float* __restrict__ out, float* __restrict__ cum,
           float* __restrict__ ess_part, int t) {
  int b = blockIdx.x, tid = threadIdx.x;
  __shared__ float s[256];
  const float* pr = out + PR_OFF + (size_t)(b * NT + (t - 1)) * NP;
  float loc[16], run = 0.f, sq = 0.f;
  for (int i = 0; i < 16; ++i) {
    float v = pr[tid * 16 + i];
    run += v; loc[i] = run; sq += v * v;
  }
  s[tid] = run; __syncthreads();
  for (int off = 1; off < 256; off <<= 1) {
    float v = (tid >= off) ? s[tid - off] : 0.f;
    __syncthreads();
    s[tid] += v;
    __syncthreads();
  }
  float excl = s[tid] - run;
  float* cb = cum + (size_t)b * NP;
  for (int i = 0; i < 16; ++i) cb[tid * 16 + i] = excl + loc[i];
  __syncthreads();
  s[tid] = sq; __syncthreads();
  for (int off = 128; off > 0; off >>= 1) {
    if (tid < off) s[tid] += s[tid + off];
    __syncthreads();
  }
  if (tid == 0) ess_part[b] = 1.f / s[0];
}

// adaptive systematic resample + motion update
__global__ __launch_bounds__(256)
void k_resample(float* __restrict__ out, const float* __restrict__ cum,
                const float* __restrict__ ess_part, const float* __restrict__ action,
                const float* __restrict__ mnz, const float* __restrict__ ru,
                const float* __restrict__ bias, float* __restrict__ logw, int t) {
  int b = blockIdx.x >> 4;
  int p = (blockIdx.x & 15) * 256 + threadIdx.x;
  float ess = 0.f;
  for (int i = 0; i < NB; ++i) ess += ess_part[i];
  ess *= (1.f / NB);
  bool rs = ess < 0.5f * NP;                                   // uniform across grid
  const float* prev = out + PL_OFF + ((size_t)(b * NT + t - 1) * NP) * 3;
  const float* prw  = out + PR_OFF + (size_t)(b * NT + t - 1) * NP;
  int idx = p;
  float base;
  if (rs) {
    float pos = (ru[b * NT + t] + (float)p) * (1.f / NP);
    const float* cb = cum + (size_t)b * NP;
    int lo = 0, hi = NP;
    while (lo < hi) { int mid = (lo + hi) >> 1; if (cb[mid] < pos) lo = mid + 1; else hi = mid; }
    idx = lo < NP - 1 ? lo : NP - 1;
    base = -logf((float)NP);
  } else {
    base = logf(prw[p] + 1e-16f);
  }
  const float* ab = action + (size_t)(b * NT + t - 1) * 3;
  const float* bb = bias + (size_t)(b * NT + t - 1) * 3;
  const float* mz = mnz + ((size_t)(b * NT + t) * NP + p) * 3;
  float* dst = out + PL_OFF + ((size_t)(b * NT + t) * NP + p) * 3;
  #pragma unroll
  for (int k = 0; k < 3; ++k)
    dst[k] = prev[idx * 3 + k] + ab[k] + (mz[k] * STDV + bb[k]);
  out[IL_OFF + (size_t)(b * NP + p) * NT + t] = (float)((rs ? idx : p) + NP * b);
  logw[(size_t)b * NP + p] = base;   // prior_log - propose_log cancels
}

__global__ void k_finish(const float* __restrict__ obs_acc, float* __restrict__ out) {
  if (threadIdx.x == 0 && blockIdx.x == 0) {
    float s = 0.f;
    for (int i = 0; i < NB; ++i) s += obs_acc[i];
    out[LL_OFF] = s / (float)(NB * NP);
  }
}

// ------------------------------------------------------------------
extern "C" void kernel_launch(void* const* d_in, const int* in_sizes, int n_in,
                              void* d_out, int out_size, void* d_ws, size_t ws_size,
                              hipStream_t stream) {
  const float* obs     = (const float*)d_in[0];
  const float* state   = (const float*)d_in[1];
  const float* action  = (const float*)d_in[2];
  const float* mnz     = (const float*)d_in[3];
  const float* ru      = (const float*)d_in[4];
  const float* inz     = (const float*)d_in[5];
  const float* W_enc1  = (const float*)d_in[6];
  const float* b_enc1  = (const float*)d_in[7];
  const float* W_enc2  = (const float*)d_in[8];
  const float* b_enc2  = (const float*)d_in[9];
  const float* W_pe1   = (const float*)d_in[10];
  const float* b_pe1   = (const float*)d_in[11];
  const float* W_pe2   = (const float*)d_in[12];
  const float* b_pe2   = (const float*)d_in[13];
  const float* W_no1   = (const float*)d_in[14];
  const float* b_no1   = (const float*)d_in[15];
  const float* W_no2   = (const float*)d_in[16];
  const float* b_no2   = (const float*)d_in[17];

  float* out = (float*)d_out;
  float* W   = (float*)d_ws;
  float* logw     = W + WS_LOGW;
  float* cum      = W + WS_CUM;
  float* bias     = W + WS_BIAS;
  float* enc_all  = W + WS_ENC;
  float* enc_norm = W + WS_ENRM;
  float* ess_part = W + WS_ESS;
  float* obs_acc  = W + WS_ACC;
  half_t* W2h     = (half_t*)(W + WS_W2H);
  float* henc     = W + WS_LOGW;  // alias: only live before k_init

  k_cvt_w2<<<(NH * NH + 255) / 256, 256, 0, stream>>>(W_pe2, W2h);
  k_enc1<<<(NB * NT * NH + 255) / 256, 256, 0, stream>>>(obs, W_enc1, b_enc1, henc);
  k_enc2<<<(NB * NT * NH + 255) / 256, 256, 0, stream>>>(henc, W_enc2, b_enc2, enc_all);
  k_norm<<<(NB * NT + 255) / 256, 256, 0, stream>>>(enc_all, enc_norm);
  k_bias<<<(NB * NT + 255) / 256, 256, 0, stream>>>(action, W_no1, b_no1, W_no2, b_no2, bias);
  k_init<<<(NB * NP + 255) / 256, 256, 0, stream>>>(state, inz, out, logw, obs_acc);

  const size_t shmem = LDS_TOTAL;

  k_measure<<<NB * 32, 256, shmem, stream>>>(out, 0, enc_all, enc_norm,
                                             W_pe1, b_pe1, W2h, b_pe2, logw);
  k_softmax<<<NB, 256, 0, stream>>>(logw, out, obs_acc, 0);

  for (int t = 1; t < NT; ++t) {
    k_cum<<<NB, 256, 0, stream>>>(out, cum, ess_part, t);
    k_resample<<<NB * 16, 256, 0, stream>>>(out, cum, ess_part, action, mnz, ru,
                                            bias, logw, t);
    k_measure<<<NB * 32, 256, shmem, stream>>>(out, t, enc_all, enc_norm,
                                               W_pe1, b_pe1, W2h, b_pe2, logw);
    k_softmax<<<NB, 256, 0, stream>>>(logw, out, obs_acc, t);
  }
  k_finish<<<1, 32, 0, stream>>>(obs_acc, out);
}
